// PotGNN_48378511622172
// MI455X (gfx1250) — matrix operand order repaired and checked by reading, not compiled, not run on hardware
//
#include <hip/hip_runtime.h>
#include <hip/hip_bf16.h>
#include <math.h>

// ---------------------------------------------------------------------------
// PotGNN on MI455X (gfx1250): bf16 WMMA GEMMs + gather-fused LDS staging.
//
// Input layout (setup_inputs dict order; params flattened in jax pytree order,
// i.e. dict keys sorted alphabetically, lists kept in order):
//   0 atomic_numbers  1 distances  2 unit_vectors  3 i  4 j
//   5 idx_i  6 idx_j  7 idx_k  8 idx_ji  9 idx_kj
//   10.. params:
//     edge_blocks[l] (l=0,1), base EB=10+12*l:
//       +0 bn_c2.b  +1 bn_c2.g  +2 bn_c2_2.b +3 bn_c2_2.g
//       +4 bn_c3.b  +5 bn_c3.g  +6 bn_c3_2.b +7 bn_c3_2.g
//       +8 lin_c2.b +9 lin_c2.w +10 lin_c3.b +11 lin_c3.w
//     34 emb   35 ne_lin1.b 36 ne_lin1.w 37 ne_lin2.b 38 ne_lin2.w
//     node_blocks[l], base NB=39+6*l:
//       +0 bn.b +1 bn.g +2 bn_c1.b +3 bn_c1.g +4 lin_c1.b +5 lin_c1.w
//     51..56 node_pol_pred (b,w)x3    57..62 pol_pred (b,w)x3
// ---------------------------------------------------------------------------

typedef __bf16 bf16;
typedef __attribute__((ext_vector_type(16))) __bf16 v16bf;
typedef __attribute__((ext_vector_type(8)))  __bf16 v8bf;
typedef __attribute__((ext_vector_type(8)))  float  v8f;

#define NATOM 10000
#define NEDGE 320000
#define HN 64
#define EPS_BN 1e-5f
#define LN2F 0.69314718055994530942f

__device__ __forceinline__ float sspf(float x) {
  // softplus(x) - ln2, numerically stable
  return fmaxf(x, 0.0f) + log1pf(__expf(-fabsf(x))) - LN2F;
}
__device__ __forceinline__ float sigm(float x) { return 1.0f / (1.0f + __expf(-x)); }

// ------------------------- WMMA tile core ----------------------------------
// Xs: LDS, 64 rows x (K+8) bf16. Wt: global, N x K bf16 (row n = W column n).
// Each of 4 waves owns 16 rows, loops over N/16 column tiles, K/32 k-steps.
// Full-tile blocks (the overwhelmingly common case: NEDGE % 64 == 0) take a
// uniform scalar-branch fast path with unguarded stores.
template <int K, int N, int ACT>
__device__ __forceinline__ void wmma_tiles(const bf16* Xs,
                                           const bf16* __restrict__ Wt,
                                           const float* __restrict__ bias,
                                           float* __restrict__ Y, int ldy,
                                           int Nout, int r0, int Rrows) {
  constexpr int LDX = K + 8;
  const int lane = threadIdx.x & 31;
  const int wave = threadIdx.x >> 5;
  const int m0 = wave * 16;
  const int g = lane >> 4;   // half-wave group
  const int mn = lane & 15;  // row (A) / col (B/C/D) within tile
  const bf16* xrow = Xs + (size_t)(m0 + mn) * LDX;
  const bool full = (r0 + 64 <= Rrows);  // uniform across block
#pragma unroll
  for (int nt = 0; nt < N / 16; ++nt) {
    v8f acc = {0.f, 0.f, 0.f, 0.f, 0.f, 0.f, 0.f, 0.f};
    const bf16* wrow = Wt + (size_t)(nt * 16 + mn) * K + 8 * g;
#pragma unroll
    for (int kt = 0; kt < K / 32; ++kt) {
      // A 16x32 bf16: lane group g covers K = kb+8g..kb+8g+7 and kb+16+8g..+7
      union { v16bf v; v8bf h[2]; } Af, Bf;
      Af.h[0] = *(const v8bf*)(xrow + kt * 32 + 8 * g);
      Af.h[1] = *(const v8bf*)(xrow + kt * 32 + 16 + 8 * g);
      // B 32x16 bf16: lane = column n, group g covers K = kb+8g..kb+8g+7 and kb+16+8g..+7
      Bf.h[0] = *(const v8bf*)(wrow + kt * 32);
      Bf.h[1] = *(const v8bf*)(wrow + kt * 32 + 16);
      acc = __builtin_amdgcn_wmma_f32_16x16x32_bf16(false, Af.v, false, Bf.v,
                                                    (short)0, acc, false, false);
    }
    const int ncol = nt * 16 + mn;
    const bool colok = (ncol < Nout);
    const float bv = colok ? bias[ncol] : 0.f;
    if (full) {
      if (colok) {
        // D layout: VGPR v holds row v + 8*g of the tile
        float* yp = Y + (long long)(r0 + m0 + 8 * g) * ldy + ncol;
#pragma unroll
        for (int v = 0; v < 8; ++v) {
          float y = acc[v] + bv;
          if (ACT == 1) y = sspf(y);
          yp[(long long)v * ldy] = y;
        }
      }
    } else {
      if (colok) {
#pragma unroll
        for (int v = 0; v < 8; ++v) {
          const int row = r0 + m0 + v + 8 * g;
          if (row < Rrows) {
            float y = acc[v] + bv;
            if (ACT == 1) y = sspf(y);
            Y[(long long)row * ldy + ncol] = y;
          }
        }
      }
    }
  }
}

// ------------------------- GEMM kernels ------------------------------------
template <int K, int N, int ACT>
__global__ void __launch_bounds__(128)
k_gemm(const float* __restrict__ X, const bf16* __restrict__ Wt,
       const float* __restrict__ bias, float* __restrict__ Y, int Rrows,
       int Nout) {
  constexpr int LDX = K + 8;
  __shared__ bf16 Xs[64 * LDX];
  const int r0 = blockIdx.x * 64;
  for (int idx = threadIdx.x; idx < 64 * (K / 4); idx += 128) {
    const int r = idx / (K / 4), q = idx % (K / 4);
    float4 f = make_float4(0.f, 0.f, 0.f, 0.f);
    if (r0 + r < Rrows) f = *(const float4*)(X + (long long)(r0 + r) * K + q * 4);
    bf16* d = Xs + r * LDX + q * 4;
    d[0] = (bf16)f.x; d[1] = (bf16)f.y; d[2] = (bf16)f.z; d[3] = (bf16)f.w;
  }
  __syncthreads();
  wmma_tiles<K, N, ACT>(Xs, Wt, bias, Y, Nout, Nout, r0, Rrows);
}

// c1: X[e] = concat(ne[i[e]], ee[e]), K=128, N=128
__global__ void __launch_bounds__(128)
k_gemm_c1(const float* __restrict__ ne, const float* __restrict__ ee,
          const int* __restrict__ ii, const bf16* __restrict__ Wt,
          const float* __restrict__ bias, float* __restrict__ Y, int Rrows) {
  constexpr int LDX = 128 + 8;
  __shared__ bf16 Xs[64 * LDX];
  const int r0 = blockIdx.x * 64;
  for (int idx = threadIdx.x; idx < 64 * 32; idx += 128) {
    const int r = idx >> 5, q = idx & 31;
    const int e = r0 + r;
    float4 f = make_float4(0.f, 0.f, 0.f, 0.f);
    if (e < Rrows) {
      if (q < 16) f = *(const float4*)(ne + (long long)ii[e] * HN + q * 4);
      else        f = *(const float4*)(ee + (long long)e * HN + (q - 16) * 4);
    }
    bf16* d = Xs + r * LDX + q * 4;
    d[0] = (bf16)f.x; d[1] = (bf16)f.y; d[2] = (bf16)f.z; d[3] = (bf16)f.w;
  }
  __syncthreads();
  wmma_tiles<128, 128, 0>(Xs, Wt, bias, Y, 128, 128, r0, Rrows);
}

// c2: X[e] = ne[i[e]] * ne[j[e]], K=64, N=128
__global__ void __launch_bounds__(128)
k_gemm_c2(const float* __restrict__ ne, const int* __restrict__ ii,
          const int* __restrict__ jj, const bf16* __restrict__ Wt,
          const float* __restrict__ bias, float* __restrict__ Y, int Rrows) {
  constexpr int LDX = 64 + 8;
  __shared__ bf16 Xs[64 * LDX];
  const int r0 = blockIdx.x * 64;
  for (int idx = threadIdx.x; idx < 64 * 16; idx += 128) {
    const int r = idx >> 4, q = idx & 15;
    const int e = r0 + r;
    float4 f = make_float4(0.f, 0.f, 0.f, 0.f);
    if (e < Rrows) {
      const float4 a = *(const float4*)(ne + (long long)ii[e] * HN + q * 4);
      const float4 b = *(const float4*)(ne + (long long)jj[e] * HN + q * 4);
      f = make_float4(a.x * b.x, a.y * b.y, a.z * b.z, a.w * b.w);
    }
    bf16* d = Xs + r * LDX + q * 4;
    d[0] = (bf16)f.x; d[1] = (bf16)f.y; d[2] = (bf16)f.z; d[3] = (bf16)f.w;
  }
  __syncthreads();
  wmma_tiles<64, 128, 0>(Xs, Wt, bias, Y, 128, 128, r0, Rrows);
}

// c3: X[t] = concat(ne[idx_i],ne[idx_j],ne[idx_k],ee[idx_ji],ee[idx_kj]), K=320
__global__ void __launch_bounds__(128)
k_gemm_c3(const float* __restrict__ ne, const float* __restrict__ ee,
          const int* __restrict__ ti, const int* __restrict__ tj,
          const int* __restrict__ tk, const int* __restrict__ tji,
          const int* __restrict__ tkj, const bf16* __restrict__ Wt,
          const float* __restrict__ bias, float* __restrict__ Y, int Rrows) {
  constexpr int LDX = 320 + 8;
  __shared__ bf16 Xs[64 * LDX];  // 41,984 B (well under 320 KB WGP LDS)
  const int r0 = blockIdx.x * 64;
  for (int idx = threadIdx.x; idx < 64 * 80; idx += 128) {
    const int r = idx / 80, q = idx % 80;
    const int e = r0 + r;
    float4 f = make_float4(0.f, 0.f, 0.f, 0.f);
    if (e < Rrows) {
      if      (q < 16) f = *(const float4*)(ne + (long long)ti[e]  * HN + q * 4);
      else if (q < 32) f = *(const float4*)(ne + (long long)tj[e]  * HN + (q - 16) * 4);
      else if (q < 48) f = *(const float4*)(ne + (long long)tk[e]  * HN + (q - 32) * 4);
      else if (q < 64) f = *(const float4*)(ee + (long long)tji[e] * HN + (q - 48) * 4);
      else             f = *(const float4*)(ee + (long long)tkj[e] * HN + (q - 64) * 4);
    }
    bf16* d = Xs + r * LDX + q * 4;
    d[0] = (bf16)f.x; d[1] = (bf16)f.y; d[2] = (bf16)f.z; d[3] = (bf16)f.w;
  }
  __syncthreads();
  wmma_tiles<320, 128, 0>(Xs, Wt, bias, Y, 128, 128, r0, Rrows);
}

// ------------------------- small / elementwise kernels ----------------------
__global__ void k_fill(float* __restrict__ p, long long n, float v) {
  for (long long i = (long long)blockIdx.x * blockDim.x + threadIdx.x; i < n;
       i += (long long)gridDim.x * blockDim.x)
    p[i] = v;
}

// W (K x N f32, row-major) -> Wt (Npad x K bf16), zero-padded columns
__global__ void k_w2bf(const float* __restrict__ W, bf16* __restrict__ Wt,
                       int K, int N, int Npad) {
  const int idx = blockIdx.x * blockDim.x + threadIdx.x;
  if (idx < Npad * K) {
    const int n = idx / K, k = idx % K;
    Wt[idx] = (bf16)(n < N ? W[(long long)k * N + n] : 0.f);
  }
}

__global__ void k_embed(const int* __restrict__ an, const float* __restrict__ emb,
                        float* __restrict__ out, long long n) {
  for (long long i = (long long)blockIdx.x * blockDim.x + threadIdx.x; i < n;
       i += (long long)gridDim.x * blockDim.x) {
    const long long a = i >> 6; const int c = (int)(i & 63);
    out[i] = sspf(emb[(long long)an[a] * HN + c]);
  }
}

__global__ void k_rbf(const float* __restrict__ dist, float* __restrict__ ee,
                      long long n) {
  const float step = 5.0f / 63.0f;
  const float coef = -0.5f / (step * step);
  for (long long i = (long long)blockIdx.x * blockDim.x + threadIdx.x; i < n;
       i += (long long)gridDim.x * blockDim.x) {
    const long long e = i >> 6; const int c = (int)(i & 63);
    const float x = dist[e] - (float)c * step;
    ee[i] = __expf(coef * x * x);
  }
}

// column sum & sum-of-squares via LDS float atomics -> global atomics
__global__ void __launch_bounds__(256)
k_colstats(const float* __restrict__ Y, long long Rrows, int C,
           float* __restrict__ sums) {
  __shared__ float ls[256];
  ls[threadIdx.x] = 0.f;
  __syncthreads();
  const long long total = Rrows * (long long)C;
  for (long long idx = (long long)blockIdx.x * blockDim.x + threadIdx.x;
       idx < total; idx += (long long)gridDim.x * blockDim.x) {
    const float v = Y[idx];
    const int c = (int)(idx % C);
    atomicAdd(&ls[c], v);
    atomicAdd(&ls[128 + c], v * v);
  }
  __syncthreads();
  if ((int)threadIdx.x < C) {
    atomicAdd(&sums[threadIdx.x], ls[threadIdx.x]);
    atomicAdd(&sums[C + threadIdx.x], ls[128 + threadIdx.x]);
  }
}

// fold BN into y*s + t: s = g*rstd, t = b - mean*s
__global__ void k_bn_fin(const float* __restrict__ sums, float invR,
                         const float* __restrict__ g, const float* __restrict__ b,
                         float* __restrict__ st, int C) {
  const int c = threadIdx.x;
  if (c < C) {
    const float mean = sums[c] * invR;
    const float var = sums[C + c] * invR - mean * mean;
    const float s = g[c] * rsqrtf(var + EPS_BN);
    st[c] = s;
    st[C + c] = b[c] - mean * s;
  }
}

// sigmoid(f)*tanh(g) of BN(Y) halves, scattered-summed into agg[seg[e]]
__global__ void __launch_bounds__(256)
k_gate_segsum(const float* __restrict__ Y, const float* __restrict__ st,
              const int* __restrict__ seg, float* __restrict__ agg,
              long long Rrows) {
  const long long total = Rrows * 64;
  for (long long idx = (long long)blockIdx.x * blockDim.x + threadIdx.x;
       idx < total; idx += (long long)gridDim.x * blockDim.x) {
    const long long e = idx >> 6; const int c = (int)(idx & 63);
    const float f = Y[e * 128 + c] * st[c] + st[128 + c];
    const float h = Y[e * 128 + 64 + c] * st[64 + c] + st[192 + c];
    atomicAdd(&agg[(long long)seg[e] * 64 + c], sigm(f) * tanhf(h));
  }
}

__global__ void __launch_bounds__(256)
k_gate(const float* __restrict__ Y, const float* __restrict__ st,
       float* __restrict__ out, long long Rrows) {
  const long long total = Rrows * 64;
  for (long long idx = (long long)blockIdx.x * blockDim.x + threadIdx.x;
       idx < total; idx += (long long)gridDim.x * blockDim.x) {
    const long long e = idx >> 6; const int c = (int)(idx & 63);
    const float f = Y[e * 128 + c] * st[c] + st[128 + c];
    const float h = Y[e * 128 + 64 + c] * st[64 + c] + st[192 + c];
    out[idx] = sigm(f) * tanhf(h);
  }
}

__global__ void __launch_bounds__(256)
k_node_update(float* __restrict__ ne, const float* __restrict__ agg,
              const float* __restrict__ st, long long n) {
  for (long long idx = (long long)blockIdx.x * blockDim.x + threadIdx.x; idx < n;
       idx += (long long)gridDim.x * blockDim.x) {
    const int c = (int)(idx & 63);
    ne[idx] = tanhf(ne[idx] + agg[idx] * st[c] + st[64 + c]);
  }
}

__global__ void __launch_bounds__(256)
k_edge_update(float* __restrict__ ee, const float* __restrict__ c2g,
              const float* __restrict__ a3, const float* __restrict__ stA,
              const float* __restrict__ stB, long long n) {
  for (long long idx = (long long)blockIdx.x * blockDim.x + threadIdx.x; idx < n;
       idx += (long long)gridDim.x * blockDim.x) {
    const int c = (int)(idx & 63);
    const float u = c2g[idx] * stA[c] + stA[64 + c];
    const float w = a3[idx] * stB[c] + stB[64 + c];
    ee[idx] = tanhf(ee[idx] + u + w);
  }
}

// Rodrigues rotation of diag(p0,p1,p1); accumulate 6 upper-tri entries.
__global__ void __launch_bounds__(256)
k_edge_pol(const float* __restrict__ ep, const float* __restrict__ uv,
           float* __restrict__ acc, long long Rrows) {
  float part[6] = {0.f, 0.f, 0.f, 0.f, 0.f, 0.f};
  for (long long e = (long long)blockIdx.x * blockDim.x + threadIdx.x; e < Rrows;
       e += (long long)gridDim.x * blockDim.x) {
    const float p0 = ep[2 * e], p1 = ep[2 * e + 1];
    const float bx = uv[3 * e], by = uv[3 * e + 1], bz = uv[3 * e + 2];
    // v = cross(e1, b) = (0, -bz, by)
    const float S[3][3] = {{0.f, -by, -bz}, {by, 0.f, 0.f}, {bz, 0.f, 0.f}};
    float S2[3][3];
#pragma unroll
    for (int a = 0; a < 3; ++a)
#pragma unroll
      for (int b = 0; b < 3; ++b)
        S2[a][b] = S[a][0] * S[0][b] + S[a][1] * S[1][b] + S[a][2] * S[2][b];
    const float q = 1.0f / (1.0f + bx);
    float Rm[3][3];
#pragma unroll
    for (int a = 0; a < 3; ++a)
#pragma unroll
      for (int b = 0; b < 3; ++b)
        Rm[a][b] = (a == b ? 1.0f : 0.0f) + S[a][b] + S2[a][b] * q;
    const float pd[3] = {p0, p1, p1};
    float A[3][3];
#pragma unroll
    for (int a = 0; a < 3; ++a)
#pragma unroll
      for (int b = 0; b < 3; ++b) A[a][b] = Rm[a][b] * pd[b];
    // inv(Rm) via adjugate
    const float c00 = Rm[1][1] * Rm[2][2] - Rm[1][2] * Rm[2][1];
    const float c01 = Rm[1][2] * Rm[2][0] - Rm[1][0] * Rm[2][2];
    const float c02 = Rm[1][0] * Rm[2][1] - Rm[1][1] * Rm[2][0];
    const float id = 1.0f / (Rm[0][0] * c00 + Rm[0][1] * c01 + Rm[0][2] * c02);
    float inv[3][3];
    inv[0][0] = c00 * id;
    inv[0][1] = (Rm[0][2] * Rm[2][1] - Rm[0][1] * Rm[2][2]) * id;
    inv[0][2] = (Rm[0][1] * Rm[1][2] - Rm[0][2] * Rm[1][1]) * id;
    inv[1][0] = c01 * id;
    inv[1][1] = (Rm[0][0] * Rm[2][2] - Rm[0][2] * Rm[2][0]) * id;
    inv[1][2] = (Rm[0][2] * Rm[1][0] - Rm[0][0] * Rm[1][2]) * id;
    inv[2][0] = c02 * id;
    inv[2][1] = (Rm[0][1] * Rm[2][0] - Rm[0][0] * Rm[2][1]) * id;
    inv[2][2] = (Rm[0][0] * Rm[1][1] - Rm[0][1] * Rm[1][0]) * id;
    float M[3][3];
#pragma unroll
    for (int a = 0; a < 3; ++a)
#pragma unroll
      for (int b = 0; b < 3; ++b)
        M[a][b] = A[a][0] * inv[0][b] + A[a][1] * inv[1][b] + A[a][2] * inv[2][b];
    part[0] += M[0][0]; part[1] += M[1][1]; part[2] += M[2][2];
    part[3] += M[0][1]; part[4] += M[0][2]; part[5] += M[1][2];
  }
  __shared__ float sh[256];
#pragma unroll
  for (int k = 0; k < 6; ++k) {
    sh[threadIdx.x] = part[k];
    __syncthreads();
    for (int s = 128; s > 0; s >>= 1) {
      if ((int)threadIdx.x < s) sh[threadIdx.x] += sh[threadIdx.x + s];
      __syncthreads();
    }
    if (threadIdx.x == 0) atomicAdd(&acc[k], sh[0]);
    __syncthreads();
  }
}

__global__ void k_np_reduce(const float* __restrict__ np,
                            float* __restrict__ acc, long long n) {
  for (long long idx = (long long)blockIdx.x * blockDim.x + threadIdx.x; idx < n;
       idx += (long long)gridDim.x * blockDim.x)
    atomicAdd(&acc[(int)(idx % 3)], np[idx]);
}

__global__ void k_out(const float* __restrict__ acc6,
                      const float* __restrict__ acc3, float* __restrict__ out) {
  const int t = threadIdx.x;
  if (t < 6)
    out[t] = acc6[t] / (float)NEDGE + (t < 3 ? acc3[t] / (float)NATOM : 0.f);
}

// ---------------------------------------------------------------------------
extern "C" void kernel_launch(void* const* d_in, const int* in_sizes, int n_in,
                              void* d_out, int out_size, void* d_ws,
                              size_t ws_size, hipStream_t stream) {
  (void)in_sizes; (void)n_in; (void)out_size; (void)ws_size;
  const int* an   = (const int*)d_in[0];
  const float* dist = (const float*)d_in[1];
  const float* uv = (const float*)d_in[2];
  const int* ii  = (const int*)d_in[3];
  const int* jj  = (const int*)d_in[4];
  const int* ti  = (const int*)d_in[5];
  const int* tj  = (const int*)d_in[6];
  const int* tk  = (const int*)d_in[7];
  const int* tji = (const int*)d_in[8];
  const int* tkj = (const int*)d_in[9];
  auto F = [&](int k) -> const float* { return (const float*)d_in[k]; };

  // -------- workspace layout (256B aligned regions) --------
  size_t off = 0;
  auto takef = [&](size_t nf) -> float* {
    off = (off + 255) & ~(size_t)255;
    float* p = (float*)((char*)d_ws + off);
    off += nf * sizeof(float);
    return p;
  };
  float* f_ne  = takef((size_t)NATOM * 64);
  float* f_tA  = takef((size_t)NATOM * 64);
  float* f_tB  = takef((size_t)NATOM * 64);
  float* f_np  = takef((size_t)NATOM * 4);
  float* f_agg = takef((size_t)NATOM * 64);
  float* f_ee  = takef((size_t)NEDGE * 64);
  float* f_Y   = takef((size_t)NEDGE * 128);
  float* f_c2g = takef((size_t)NEDGE * 64);
  float* f_a3  = takef((size_t)NEDGE * 64);
  float* f_sums = takef(256);
  float* f_stA  = takef(256);
  float* f_stB  = takef(256);
  float* f_stC  = takef(256);
  float* f_acc  = takef(16);
  off = (off + 255) & ~(size_t)255;
  bf16* wtbase = (bf16*)((char*)d_ws + off);
  size_t wo = 0;
  auto takew = [&](size_t n) -> bf16* { bf16* p = wtbase + wo; wo += n; return p; };
  bf16* wt_nl1 = takew(64 * 64);
  bf16* wt_nl2 = takew(64 * 64);
  bf16* wt_c1[2] = {takew(128 * 128), takew(128 * 128)};
  bf16* wt_c2[2] = {takew(128 * 64), takew(128 * 64)};
  bf16* wt_c3[2] = {takew(128 * 320), takew(128 * 320)};
  bf16* wt_p0 = takew(64 * 64);
  bf16* wt_p1 = takew(64 * 64);
  bf16* wt_p2 = takew(16 * 64);
  bf16* wt_q0 = takew(64 * 64);
  bf16* wt_q1 = takew(64 * 64);
  bf16* wt_q2 = takew(16 * 64);

  auto w2bf = [&](const float* W, bf16* Wt, int K, int N, int Npad) {
    const int n = Npad * K;
    k_w2bf<<<(n + 255) / 256, 256, 0, stream>>>(W, Wt, K, N, Npad);
  };
  auto fill = [&](float* p, long long n, float v) {
    long long b = (n + 255) / 256; if (b > 4096) b = 4096;
    k_fill<<<(int)b, 256, 0, stream>>>(p, n, v);
  };

  // -------- weight conversion (f32 -> bf16 transposed) --------
  w2bf(F(36), wt_nl1, 64, 64, 64);
  w2bf(F(38), wt_nl2, 64, 64, 64);
  for (int l = 0; l < 2; ++l) {
    const int EB = 10 + 12 * l, NB = 39 + 6 * l;
    w2bf(F(NB + 5), wt_c1[l], 128, 128, 128);
    w2bf(F(EB + 9), wt_c2[l], 64, 128, 128);
    w2bf(F(EB + 11), wt_c3[l], 320, 128, 128);
  }
  w2bf(F(58), wt_p0, 64, 64, 64);
  w2bf(F(60), wt_p1, 64, 64, 64);
  w2bf(F(62), wt_p2, 64, 2, 16);
  w2bf(F(52), wt_q0, 64, 64, 64);
  w2bf(F(54), wt_q1, 64, 64, 64);
  w2bf(F(56), wt_q2, 64, 3, 16);

  const int GE = (NEDGE + 63) / 64;  // 5000 blocks
  const int GA = (NATOM + 63) / 64;  // 157 blocks
  const int EW = 4096;               // elementwise grid

  // -------- front: node embedding MLP + edge RBF --------
  k_embed<<<EW, 256, 0, stream>>>(an, F(34), f_tA, (long long)NATOM * 64);
  k_gemm<64, 64, 1><<<GA, 128, 0, stream>>>(f_tA, wt_nl1, F(35), f_tB, NATOM, 64);
  k_gemm<64, 64, 0><<<GA, 128, 0, stream>>>(f_tB, wt_nl2, F(37), f_ne, NATOM, 64);
  k_rbf<<<EW, 256, 0, stream>>>(dist, f_ee, (long long)NEDGE * 64);

  // -------- interaction layers --------
  for (int l = 0; l < 2; ++l) {
    const int EB = 10 + 12 * l, NB = 39 + 6 * l;
    // c1 -> agg -> node update
    fill(f_sums, 256, 0.f);
    k_gemm_c1<<<GE, 128, 0, stream>>>(f_ne, f_ee, ii, wt_c1[l], F(NB + 4), f_Y, NEDGE);
    k_colstats<<<1024, 256, 0, stream>>>(f_Y, NEDGE, 128, f_sums);
    k_bn_fin<<<1, 128, 0, stream>>>(f_sums, 1.f / NEDGE, F(NB + 3), F(NB + 2), f_stA, 128);
    fill(f_agg, (long long)NATOM * 64, 0.f);
    k_gate_segsum<<<EW, 256, 0, stream>>>(f_Y, f_stA, ii, f_agg, NEDGE);
    fill(f_sums, 256, 0.f);
    k_colstats<<<256, 256, 0, stream>>>(f_agg, NATOM, 64, f_sums);
    k_bn_fin<<<1, 128, 0, stream>>>(f_sums, 1.f / NATOM, F(NB + 1), F(NB + 0), f_stA, 64);
    k_node_update<<<EW, 256, 0, stream>>>(f_ne, f_agg, f_stA, (long long)NATOM * 64);
    // c2 -> gated -> BN2 (stats in stA, applied at edge update)
    fill(f_sums, 256, 0.f);
    k_gemm_c2<<<GE, 128, 0, stream>>>(f_ne, ii, jj, wt_c2[l], F(EB + 8), f_Y, NEDGE);
    k_colstats<<<1024, 256, 0, stream>>>(f_Y, NEDGE, 128, f_sums);
    k_bn_fin<<<1, 128, 0, stream>>>(f_sums, 1.f / NEDGE, F(EB + 1), F(EB + 0), f_stA, 128);
    k_gate<<<EW, 256, 0, stream>>>(f_Y, f_stA, f_c2g, NEDGE);
    fill(f_sums, 256, 0.f);
    k_colstats<<<1024, 256, 0, stream>>>(f_c2g, NEDGE, 64, f_sums);
    k_bn_fin<<<1, 128, 0, stream>>>(f_sums, 1.f / NEDGE, F(EB + 3), F(EB + 2), f_stA, 64);
    // c3 (triplets) -> segsum over idx_ji -> BN2 (stats in stB)
    fill(f_sums, 256, 0.f);
    k_gemm_c3<<<GE, 128, 0, stream>>>(f_ne, f_ee, ti, tj, tk, tji, tkj, wt_c3[l],
                                      F(EB + 10), f_Y, NEDGE);
    k_colstats<<<1024, 256, 0, stream>>>(f_Y, NEDGE, 128, f_sums);
    k_bn_fin<<<1, 128, 0, stream>>>(f_sums, 1.f / NEDGE, F(EB + 5), F(EB + 4), f_stC, 128);
    fill(f_a3, (long long)NEDGE * 64, 0.f);
    k_gate_segsum<<<EW, 256, 0, stream>>>(f_Y, f_stC, tji, f_a3, NEDGE);
    fill(f_sums, 256, 0.f);
    k_colstats<<<1024, 256, 0, stream>>>(f_a3, NEDGE, 64, f_sums);
    k_bn_fin<<<1, 128, 0, stream>>>(f_sums, 1.f / NEDGE, F(EB + 7), F(EB + 6), f_stB, 64);
    k_edge_update<<<EW, 256, 0, stream>>>(f_ee, f_c2g, f_a3, f_stA, f_stB,
                                          (long long)NEDGE * 64);
  }

  // -------- edge polarizability head --------
  k_gemm<64, 64, 1><<<GE, 128, 0, stream>>>(f_ee, wt_p0, F(57), f_a3, NEDGE, 64);
  k_gemm<64, 64, 1><<<GE, 128, 0, stream>>>(f_a3, wt_p1, F(59), f_c2g, NEDGE, 64);
  k_gemm<64, 16, 0><<<GE, 128, 0, stream>>>(f_c2g, wt_p2, F(61), f_Y, NEDGE, 2);
  fill(f_acc, 16, 0.f);
  k_edge_pol<<<2048, 256, 0, stream>>>(f_Y, uv, f_acc, NEDGE);

  // -------- node polarizability head --------
  k_gemm<64, 64, 1><<<GA, 128, 0, stream>>>(f_ne, wt_q0, F(51), f_tA, NATOM, 64);
  k_gemm<64, 64, 1><<<GA, 128, 0, stream>>>(f_tA, wt_q1, F(53), f_tB, NATOM, 64);
  k_gemm<64, 16, 0><<<GA, 128, 0, stream>>>(f_tB, wt_q2, F(55), f_np, NATOM, 3);
  k_np_reduce<<<128, 256, 0, stream>>>(f_np, f_acc + 8, (long long)NATOM * 3);

  k_out<<<1, 32, 0, stream>>>(f_acc, f_acc + 8, (float*)d_out);
}
